// GRUCell_89515708383962
// MI455X (gfx1250) — matrix operand order repaired and checked by reading, not compiled, not run on hardware
//
#include <hip/hip_runtime.h>

// ---------------------------------------------------------------------------
// GRU cell with h_prev == 0 (reference hard-codes h_prev = zeros):
//   z = sigmoid(x @ w_iz^T), n = tanh(x @ w_in^T), out = (1 - z) * n
// => two fused GEMMs (8192x2048)@(2048x2048)^T + elementwise epilogue.
// weight_hh and w_ir are mathematically dead.
// Precision: split-bf16 (hi/lo), A*B ~= Ahi*Bhi + Alo*Bhi + Ahi*Blo.
//
// Fast path (needs ~100.7MB workspace):
//   1) pack_hilo: one-shot split of x / w_iz / w_in into bf16 hi+lo planes.
//   2) gru_wmma_async: hot loop moves pre-split bf16 tiles with
//      GLOBAL_LOAD_ASYNC_TO_LDS_B128 (ASYNCcnt) -> no staging regs, no
//      conversion VALU, no ds_stores -> <256 VGPRs (no s_set_vgpr_msb churn).
// Fallback (small ws): round-2 self-contained kernel (inline conversion).
// ---------------------------------------------------------------------------

#define MDIM 8192
#define KDIM 2048
#define HID  2048

#define BM 128
#define BN 128
#define BK 32
#define LDS_STRIDE 40                            // padded bf16 per LDS row
#define A_PLANE (BM * LDS_STRIDE)                // 5120 bf16
#define B_PLANE (BN * LDS_STRIDE)                // 5120 bf16
#define BUF_HALF (2 * A_PLANE + 4 * B_PLANE)     // 30720 bf16 per buffer
#define BUF_BYTES (BUF_HALF * 2)                 // 61440 B
#define NKT (KDIM / BK)                          // 64 K-steps

// workspace layout (bytes)
#define WS_XH  ((size_t)0)
#define WS_XL  ((size_t)33554432)
#define WS_WZH ((size_t)67108864)
#define WS_WZL ((size_t)75497472)
#define WS_WNH ((size_t)83886080)
#define WS_WNL ((size_t)92274688)
#define WS_NEED ((size_t)100663296)

typedef __attribute__((ext_vector_type(16))) __bf16 v16bf;
typedef __attribute__((ext_vector_type(8)))  __bf16 bf16x8;
typedef __attribute__((ext_vector_type(4)))  __bf16 bf16x4;
typedef __attribute__((ext_vector_type(8)))  float  v8f;
typedef __attribute__((ext_vector_type(4)))  float  f32x4;
typedef int i32x4 __attribute__((vector_size(16)));   // matches builtin param

union Frag { v16bf v; bf16x8 h[2]; };
union F4   { f32x4 v; float  e[4]; };
union B4   { bf16x4 v; __bf16 e[4]; };
union F8   { v8f   v; float  e[8]; };

#define WMMA_BF16(A, B, C) \
  __builtin_amdgcn_wmma_f32_16x16x32_bf16(false, (A), false, (B), (short)0, (C), false, false)

__device__ __forceinline__ void cvt_hilo(f32x4 v, bf16x4& h, bf16x4& l) {
  F4 u; u.v = v;
  B4 uh, ul;
#pragma unroll
  for (int j = 0; j < 4; ++j) {
    __bf16 hb = (__bf16)u.e[j];
    uh.e[j] = hb;
    ul.e[j] = (__bf16)(u.e[j] - (float)hb);
  }
  h = uh.v; l = ul.v;
}

// ---- CDNA5 async copy helpers (builtin confirmed present by round-3 probe) -
__device__ __forceinline__ void async_b128(const void* g, unsigned lds_byte_off) {
#if __has_builtin(__builtin_amdgcn_global_load_async_to_lds_b128)
  __builtin_amdgcn_global_load_async_to_lds_b128(
      (__attribute__((address_space(1))) i32x4*)(unsigned long long)g,
      (__attribute__((address_space(3))) i32x4*)(unsigned long long)lds_byte_off,
      0, 0);
#else
  asm volatile("global_load_async_to_lds_b128 %0, %1, off"
               :: "v"(lds_byte_off), "v"(g) : "memory");
#endif
}

__device__ __forceinline__ void wait_async_zero() {
#if __has_builtin(__builtin_amdgcn_s_wait_asynccnt)
  __builtin_amdgcn_s_wait_asynccnt(0);
#else
  asm volatile("s_wait_asynccnt 0x0" ::: "memory");
#endif
}

// ---------------------------------------------------------------------------
// One-shot f32 -> (bf16 hi, bf16 lo) split, 4 floats / thread.
// ---------------------------------------------------------------------------
__global__ void __launch_bounds__(256)
pack_hilo(const float* __restrict__ src, __bf16* __restrict__ hi,
          __bf16* __restrict__ lo, int n4) {
  const int i = blockIdx.x * 256 + threadIdx.x;
  if (i < n4) {
    f32x4 v = ((const f32x4*)src)[i];
    bf16x4 h, l;
    cvt_hilo(v, h, l);
    ((bf16x4*)hi)[i] = h;
    ((bf16x4*)lo)[i] = l;
  }
}

// ---------------------------------------------------------------------------
// Fast-path GEMM: pre-split bf16 planes, async global->LDS staging.
// Block 128M x 128N(x2 gates); 8 waves = 2(M) x 4(N); wave = 64M x 32N.
// ---------------------------------------------------------------------------
__global__ void __launch_bounds__(256)
gru_wmma_async(const __bf16* __restrict__ xh, const __bf16* __restrict__ xl,
               const __bf16* __restrict__ wzh, const __bf16* __restrict__ wzl,
               const __bf16* __restrict__ wnh, const __bf16* __restrict__ wnl,
               float* __restrict__ out)
{
  extern __shared__ __align__(16) char smem_raw[];
  __bf16* const smem = reinterpret_cast<__bf16*>(smem_raw);
  const unsigned sbase = (unsigned)(unsigned long long)(void*)smem; // LDS byte offset

  const int tid  = threadIdx.x;
  const int lane = tid & 31;
  const int wave = tid >> 5;
  const int wm   = wave & 1;       // 2 waves along M (64 rows each)
  const int wn   = wave >> 1;      // 4 waves along N (32 cols each)
  const int lr   = lane & 15;
  const int lh   = lane >> 4;
  const int m0   = blockIdx.y * BM;
  const int n0   = blockIdx.x * BN;

  // 12 async b128 chunks / thread / K-step:
  //   i 0..1 : A-hi   i 2..3 : A-lo   i 4..11 : B planes (zh, zl, nh, nl)
  // chunk f: row = f>>2 (0..127), c = f&3 -> 8 bf16 at col c*8.
  const __bf16* gsrc[12];
  unsigned loff[12];                // LDS byte offset within buffer
#pragma unroll
  for (int i = 0; i < 12; ++i) {
    const __bf16* plane;
    int planeBase, r0;
    int ii;
    if (i < 4) { plane = (i < 2) ? xh : xl; planeBase = (i < 2) ? 0 : A_PLANE;
                 r0 = m0; ii = i & 1; }
    else {
      const int q = (i - 4) >> 1;   // 0:zh 1:zl 2:nh 3:nl
      plane = (q == 0) ? wzh : (q == 1) ? wzl : (q == 2) ? wnh : wnl;
      planeBase = 2 * A_PLANE + q * B_PLANE;
      r0 = n0; ii = (i - 4) & 1;
    }
    const int f   = tid + ii * 256;
    const int row = f >> 2;
    const int c   = f & 3;
    gsrc[i] = plane + (size_t)(r0 + row) * KDIM + c * 8;
    loff[i] = (unsigned)(planeBase + row * LDS_STRIDE + c * 8) * 2u;
  }

  v8f acc[2][4][2];                 // [gate][mi][ni]
  const v8f vzero = {0.f, 0.f, 0.f, 0.f, 0.f, 0.f, 0.f, 0.f};
#pragma unroll
  for (int g = 0; g < 2; ++g)
#pragma unroll
    for (int mi = 0; mi < 4; ++mi)
#pragma unroll
      for (int ni = 0; ni < 2; ++ni) acc[g][mi][ni] = vzero;

  // ---- prologue: async-stage tile 0 into buffer 0 ----
#pragma unroll
  for (int i = 0; i < 12; ++i) async_b128(gsrc[i], sbase + loff[i]);
  wait_async_zero();
  __syncthreads();

#pragma unroll 1
  for (int kt = 0; kt < NKT; ++kt) {
    const __bf16* buf = smem + (kt & 1) * BUF_HALF;

    // Kick off async staging of the next tile into the other buffer.
    if (kt + 1 < NKT) {
      const unsigned nb = sbase + ((kt + 1) & 1) * BUF_BYTES;
#pragma unroll
      for (int i = 0; i < 12; ++i)
        async_b128(gsrc[i] + (kt + 1) * BK, nb + loff[i]);
    }

    // ---- LDS -> B fragments (kept live across the mi loop) ----
    Frag bfr[2][2][2];              // [gate][ni][hi/lo]
#pragma unroll
    for (int g = 0; g < 2; ++g)
#pragma unroll
      for (int ni = 0; ni < 2; ++ni) {
        const int row = wn * 32 + ni * 16 + lr;
#pragma unroll
        for (int p = 0; p < 2; ++p) {
          const __bf16* pB = buf + 2 * A_PLANE + (2 * g + p) * B_PLANE
                           + row * LDS_STRIDE + lh * 8;
          bfr[g][ni][p].h[0] = *(const bf16x8*)(pB);
          bfr[g][ni][p].h[1] = *(const bf16x8*)(pB + 16);
        }
      }

    // ---- mi loop: 2 A-frag loads feed 12 WMMAs each (48 WMMA / K-step) ----
#pragma unroll
    for (int mi = 0; mi < 4; ++mi) {
      const int row = wm * 64 + mi * 16 + lr;
      const __bf16* pH = buf + row * LDS_STRIDE + lh * 8;
      const __bf16* pL = pH + A_PLANE;
      Frag aH, aL;
      aH.h[0] = *(const bf16x8*)(pH);
      aH.h[1] = *(const bf16x8*)(pH + 16);
      aL.h[0] = *(const bf16x8*)(pL);
      aL.h[1] = *(const bf16x8*)(pL + 16);
#pragma unroll
      for (int g = 0; g < 2; ++g)
#pragma unroll
        for (int ni = 0; ni < 2; ++ni) {
          v8f c = acc[g][mi][ni];
          c = WMMA_BF16(aH.v, bfr[g][ni][0].v, c);
          c = WMMA_BF16(aL.v, bfr[g][ni][0].v, c);
          c = WMMA_BF16(aH.v, bfr[g][ni][1].v, c);
          acc[g][mi][ni] = c;
        }
    }

    if (kt + 1 < NKT) {
      wait_async_zero();            // my wave's async stores into nbuf landed
      __syncthreads();              // everyone's landed; prev reads done
    }
  }

  // ---- fused epilogue: out = (1 - sigmoid(z)) * tanh(n) ----
#pragma unroll
  for (int mi = 0; mi < 4; ++mi)
#pragma unroll
    for (int ni = 0; ni < 2; ++ni) {
      const int col = n0 + wn * 32 + ni * 16 + lr;
      float* op = out + (size_t)(m0 + wm * 64 + mi * 16 + lh * 8) * HID + col;
      F8 uz, un;
      uz.v = acc[0][mi][ni];
      un.v = acc[1][mi][ni];
#pragma unroll
      for (int r = 0; r < 8; ++r) {
        const float gz = 1.0f / (1.0f + __expf(uz.e[r]));                // 1 - sigmoid(z)
        const float gn = 1.0f - 2.0f / (1.0f + __expf(2.0f * un.e[r]));  // tanh(n)
        op[(size_t)r * HID] = gz * gn;
      }
    }
}

// ---------------------------------------------------------------------------
// Fallback kernel (self-contained, inline f32->bf16 hi/lo conversion).
// ---------------------------------------------------------------------------
__global__ void __launch_bounds__(256)
gru_fused_wmma(const float* __restrict__ x,
               const float* __restrict__ w_iz,
               const float* __restrict__ w_in,
               float* __restrict__ out)
{
  extern __shared__ __align__(16) char smem_raw[];
  __bf16* const smem = reinterpret_cast<__bf16*>(smem_raw);

  const int tid  = threadIdx.x;
  const int lane = tid & 31;
  const int wave = tid >> 5;
  const int wm   = wave & 1;
  const int wn   = wave >> 1;
  const int lr   = lane & 15;
  const int lh   = lane >> 4;
  const int m0   = blockIdx.y * BM;
  const int n0   = blockIdx.x * BN;

  const float* baseA[4];
  const float* baseB[8];
  int aOff[4], bOff[8];
#pragma unroll
  for (int i = 0; i < 4; ++i) {
    const int f  = tid + i * 256;
    const int ra = f >> 3;
    const int ca = (f & 7) * 4;
    baseA[i] = x + (size_t)(m0 + ra) * KDIM + ca;
    aOff[i]  = ra * LDS_STRIDE + ca;
  }
#pragma unroll
  for (int i = 0; i < 8; ++i) {
    const int f  = tid + i * 256;
    const int rb = (f >> 3) & 127;
    const int ca = (f & 7) * 4;
    baseB[i] = ((i >> 2) ? w_in : w_iz) + (size_t)(n0 + rb) * KDIM + ca;
    bOff[i]  = rb * LDS_STRIDE + ca;
  }

  v8f acc[2][4][2];
  const v8f vzero = {0.f, 0.f, 0.f, 0.f, 0.f, 0.f, 0.f, 0.f};
#pragma unroll
  for (int g = 0; g < 2; ++g)
#pragma unroll
    for (int mi = 0; mi < 4; ++mi)
#pragma unroll
      for (int ni = 0; ni < 2; ++ni) acc[g][mi][ni] = vzero;

  f32x4 aR[4], bR[8];
#pragma unroll
  for (int i = 0; i < 4; ++i) aR[i] = *(const f32x4*)(baseA[i]);
#pragma unroll
  for (int i = 0; i < 8; ++i) bR[i] = *(const f32x4*)(baseB[i]);
  {
    __bf16* buf = smem;
#pragma unroll
    for (int i = 0; i < 4; ++i) {
      bf16x4 h, l;
      cvt_hilo(aR[i], h, l);
      *(bf16x4*)(buf + aOff[i])           = h;
      *(bf16x4*)(buf + A_PLANE + aOff[i]) = l;
    }
#pragma unroll
    for (int i = 0; i < 8; ++i) {
      bf16x4 h, l;
      cvt_hilo(bR[i], h, l);
      const int g = i >> 2;
      *(bf16x4*)(buf + 2 * A_PLANE + (2 * g + 0) * B_PLANE + bOff[i]) = h;
      *(bf16x4*)(buf + 2 * A_PLANE + (2 * g + 1) * B_PLANE + bOff[i]) = l;
    }
  }
  __syncthreads();

#pragma unroll 1
  for (int kt = 0; kt < NKT; ++kt) {
    const __bf16* buf = smem + (kt & 1) * BUF_HALF;
    if (kt + 1 < NKT) {
#pragma unroll
      for (int i = 0; i < 4; ++i) aR[i] = *(const f32x4*)(baseA[i] + (kt + 1) * BK);
#pragma unroll
      for (int i = 0; i < 8; ++i) bR[i] = *(const f32x4*)(baseB[i] + (kt + 1) * BK);
    }

    Frag bfr[2][2][2];
#pragma unroll
    for (int g = 0; g < 2; ++g)
#pragma unroll
      for (int ni = 0; ni < 2; ++ni) {
        const int row = wn * 32 + ni * 16 + lr;
#pragma unroll
        for (int p = 0; p < 2; ++p) {
          const __bf16* pB = buf + 2 * A_PLANE + (2 * g + p) * B_PLANE
                           + row * LDS_STRIDE + lh * 8;
          bfr[g][ni][p].h[0] = *(const bf16x8*)(pB);
          bfr[g][ni][p].h[1] = *(const bf16x8*)(pB + 16);
        }
      }

#pragma unroll
    for (int mi = 0; mi < 4; ++mi) {
      const int row = wm * 64 + mi * 16 + lr;
      const __bf16* pH = buf + row * LDS_STRIDE + lh * 8;
      const __bf16* pL = pH + A_PLANE;
      Frag aH, aL;
      aH.h[0] = *(const bf16x8*)(pH);
      aH.h[1] = *(const bf16x8*)(pH + 16);
      aL.h[0] = *(const bf16x8*)(pL);
      aL.h[1] = *(const bf16x8*)(pL + 16);
#pragma unroll
      for (int g = 0; g < 2; ++g)
#pragma unroll
        for (int ni = 0; ni < 2; ++ni) {
          v8f c = acc[g][mi][ni];
          c = WMMA_BF16(aH.v, bfr[g][ni][0].v, c);
          c = WMMA_BF16(aL.v, bfr[g][ni][0].v, c);
          c = WMMA_BF16(aH.v, bfr[g][ni][1].v, c);
          acc[g][mi][ni] = c;
        }
    }

    if (kt + 1 < NKT) {
      __bf16* nbuf = smem + ((kt + 1) & 1) * BUF_HALF;
#pragma unroll
      for (int i = 0; i < 4; ++i) {
        bf16x4 h, l;
        cvt_hilo(aR[i], h, l);
        *(bf16x4*)(nbuf + aOff[i])           = h;
        *(bf16x4*)(nbuf + A_PLANE + aOff[i]) = l;
      }
#pragma unroll
      for (int i = 0; i < 8; ++i) {
        bf16x4 h, l;
        cvt_hilo(bR[i], h, l);
        const int g = i >> 2;
        *(bf16x4*)(nbuf + 2 * A_PLANE + (2 * g + 0) * B_PLANE + bOff[i]) = h;
        *(bf16x4*)(nbuf + 2 * A_PLANE + (2 * g + 1) * B_PLANE + bOff[i]) = l;
      }
      __syncthreads();
    }
  }

#pragma unroll
  for (int mi = 0; mi < 4; ++mi)
#pragma unroll
    for (int ni = 0; ni < 2; ++ni) {
      const int col = n0 + wn * 32 + ni * 16 + lr;
      float* op = out + (size_t)(m0 + wm * 64 + mi * 16 + lh * 8) * HID + col;
      F8 uz, un;
      uz.v = acc[0][mi][ni];
      un.v = acc[1][mi][ni];
#pragma unroll
      for (int r = 0; r < 8; ++r) {
        const float gz = 1.0f / (1.0f + __expf(uz.e[r]));
        const float gn = 1.0f - 2.0f / (1.0f + __expf(2.0f * un.e[r]));
        op[(size_t)r * HID] = gz * gn;
      }
    }
}

extern "C" void kernel_launch(void* const* d_in, const int* in_sizes, int n_in,
                              void* d_out, int out_size, void* d_ws, size_t ws_size,
                              hipStream_t stream) {
  (void)in_sizes; (void)n_in; (void)out_size;
  const float* x   = (const float*)d_in[0];
  const float* wih = (const float*)d_in[1];
  // d_in[2] (weight_hh) is dead: h_prev == 0 in the reference.
  float* out = (float*)d_out;
  const float* w_iz = wih + (size_t)HID * KDIM;        // rows [H, 2H)
  const float* w_in = wih + (size_t)2 * HID * KDIM;    // rows [2H, 3H)
  dim3 grid(HID / BN, MDIM / BM);                      // 16 x 64 blocks
  const size_t shmem = (size_t)2 * BUF_BYTES;          // 122880 B dynamic LDS

  if (ws_size >= WS_NEED) {
    char* ws = (char*)d_ws;
    __bf16* xhp  = (__bf16*)(ws + WS_XH);
    __bf16* xlp  = (__bf16*)(ws + WS_XL);
    __bf16* wzhp = (__bf16*)(ws + WS_WZH);
    __bf16* wzlp = (__bf16*)(ws + WS_WZL);
    __bf16* wnhp = (__bf16*)(ws + WS_WNH);
    __bf16* wnlp = (__bf16*)(ws + WS_WNL);
    const int x4 = MDIM * KDIM / 4;                    // 4194304
    const int w4 = HID * KDIM / 4;                     // 1048576
    pack_hilo<<<dim3(x4 / 256), dim3(256), 0, stream>>>(x, xhp, xlp, x4);
    pack_hilo<<<dim3(w4 / 256), dim3(256), 0, stream>>>(w_iz, wzhp, wzlp, w4);
    pack_hilo<<<dim3(w4 / 256), dim3(256), 0, stream>>>(w_in, wnhp, wnlp, w4);
    gru_wmma_async<<<grid, dim3(256, 1, 1), shmem, stream>>>(
        xhp, xlp, wzhp, wzlp, wnhp, wnlp, out);
  } else {
    gru_fused_wmma<<<grid, dim3(256, 1, 1), shmem, stream>>>(x, w_iz, w_in, out);
  }
}
